// MeshGraphNet_2740189135778
// MI455X (gfx1250) — matrix operand order repaired and checked by reading, not compiled, not run on hardware
//
#include <hip/hip_runtime.h>

// ---------------------------------------------------------------------------
// MeshGraphNet forward for MI455X (gfx1250): bf16 activations + v_wmma bf16
// ---------------------------------------------------------------------------

typedef __bf16 bf16;
typedef __bf16 v16bf __attribute__((ext_vector_type(16)));
typedef float  v8f   __attribute__((ext_vector_type(8)));
typedef int    v4i   __attribute__((ext_vector_type(4)));

#define N_NODES 50000
#define N_EDGES 800000
#define HID 128
#define DEPTH 8

// LDS row strides (elements), padded to avoid 64-bank degenerate strides.
#define LDA_E 392   // edge A tile [128][384] -> stride 392 bf16
#define LDA_N 264   // node A tile [128][256] -> stride 264 bf16
#define LDA_D 136   // decoder A tile [128][128]
#define LDW   136   // weight tiles [K][128] -> stride 136 bf16 (also H tile)
#define LDO   132   // f32 output tile [128][128] -> stride 132 f32

// ----- async global->LDS copy (CDNA5 GLOBAL_LOAD_ASYNC_TO_LDS_B128) --------
#if defined(__has_builtin)
#if __has_builtin(__builtin_amdgcn_global_load_async_to_lds_b128)
#define MGN_HAS_ASYNC_LDS 1
#endif
#endif

__device__ __forceinline__ void g2l_b128(void* lds_dst, const void* g_src) {
#ifdef MGN_HAS_ASYNC_LDS
  __builtin_amdgcn_global_load_async_to_lds_b128(
      (__attribute__((address_space(1))) v4i*)(g_src),
      (__attribute__((address_space(3))) v4i*)(lds_dst),
      0, 0);
#else
  *(uint4*)lds_dst = *(const uint4*)g_src;
#endif
}

__device__ __forceinline__ void wait_async_lds() {
#ifdef MGN_HAS_ASYNC_LDS
#if __has_builtin(__builtin_amdgcn_s_wait_asynccnt)
  __builtin_amdgcn_s_wait_asynccnt(0);
#else
  asm volatile("s_wait_asynccnt 0" ::: "memory");
#endif
#endif
}

// ----- WMMA fragment loads (per CDNA5 ISA §7.12.2 layouts) ------------------
// A 16x32 bf16: lane L<16 -> row m0+L, K in {kk..kk+7, kk+16..kk+23}
//               lane L>=16 -> row m0+L-16, K in {kk+8..kk+15, kk+24..kk+31}
__device__ __forceinline__ v16bf load_a_frag(const bf16* A, int lda,
                                             int m0, int kk, int lane) {
  int row = m0 + (lane & 15);
  int kb  = kk + ((lane & 16) >> 1);  // +8 for lanes 16..31
  const bf16* p = A + row * lda + kb;
  union { v16bf v; uint4 q[2]; } u;
  u.q[0] = *(const uint4*)(p);
  u.q[1] = *(const uint4*)(p + 16);
  return u.v;
}

// B 32x16 bf16: lane L -> col N=n0+(L&15), K in {kk+8h..kk+8h+7, kk+16+8h..kk+23+8h}, h=L>>4
__device__ __forceinline__ v16bf load_b_frag(const bf16* B, int ldb,
                                             int kk, int n0, int lane) {
  const int kb = kk + ((lane & 16) >> 1);
  const bf16* p = B + (size_t)kb * ldb + n0 + (lane & 15);
  v16bf v;
#pragma unroll
  for (int s = 0; s < 8; ++s) {
    v[s] = p[(size_t)s * ldb];
    v[s + 8] = p[(size_t)(s + 16) * ldb];
  }
  return v;
}

// Block GEMM: D[128][128] = A[128][K] x B[K][128]; 8 waves, each wave owns a
// 16-row stripe (8 n-tiles of 16x16). EPI: 0 = +bias,relu -> bf16 out
//                                          1 = +bias       -> f32 out
//                                          2 = +bias,relu  -> f32 out
template <int K, int EPI>
__device__ __forceinline__ void block_gemm_128(const bf16* __restrict__ A, int lda,
                                               const bf16* __restrict__ B, int ldb,
                                               const float* __restrict__ bias,
                                               bf16* __restrict__ obf, int ldobf,
                                               float* __restrict__ of, int ldof) {
  const int lane = threadIdx.x & 31;
  const int m0 = (threadIdx.x >> 5) * 16;
  v8f acc[8];
#pragma unroll
  for (int n = 0; n < 8; ++n) acc[n] = (v8f)(0.0f);

#pragma unroll
  for (int kk = 0; kk < K; kk += 32) {
    v16bf a = load_a_frag(A, lda, m0, kk, lane);
#pragma unroll
    for (int n = 0; n < 8; ++n) {
      v16bf b = load_b_frag(B, ldb, kk, n * 16, lane);
      acc[n] = __builtin_amdgcn_wmma_f32_16x16x32_bf16(
          false, a, false, b, (short)0, acc[n], false, false);
    }
  }

  const int rbase = m0 + ((lane & 16) >> 1);  // +8 for lanes 16..31
#pragma unroll
  for (int n = 0; n < 8; ++n) {
    const int col = n * 16 + (lane & 15);
    const float bv = bias[col];
#pragma unroll
    for (int r = 0; r < 8; ++r) {
      const int row = rbase + r;
      float v = acc[n][r] + bv;
      if (EPI == 0) {
        v = fmaxf(v, 0.0f);
        obf[row * ldobf + col] = (bf16)v;
      } else if (EPI == 1) {
        of[row * ldof + col] = v;
      } else {
        of[row * ldof + col] = fmaxf(v, 0.0f);
      }
    }
  }
}

// Per-row LayerNorm statistics: two threads per row, 64 cols each.
__device__ __forceinline__ void row_stats(const float* __restrict__ OUT,
                                          float* __restrict__ red, int tid,
                                          float& mu, float& rs) {
  const int row = tid >> 1, half = tid & 1;
  const float* p = OUT + row * LDO + half * 64;
  float s = 0.0f, s2 = 0.0f;
#pragma unroll 8
  for (int c = 0; c < 64; ++c) {
    float v = p[c];
    s += v;
    s2 += v * v;
  }
  red[tid] = s;
  red[256 + tid] = s2;
  __syncthreads();
  const float sum = red[row * 2] + red[row * 2 + 1];
  const float sq = red[256 + row * 2] + red[256 + row * 2 + 1];
  mu = sum * (1.0f / 128.0f);
  const float var = sq * (1.0f / 128.0f) - mu * mu;
  rs = rsqrtf(var + 1e-5f);
}

__device__ __forceinline__ void stage_weight(bf16* __restrict__ dst,
                                             const bf16* __restrict__ src,
                                             int krows, int tid) {
  const int chunks = krows * 16;  // 16 x (8 bf16) chunks per row of 128
  for (int i = tid; i < chunks; i += 256) {
    const int k = i >> 4, c = i & 15;
    g2l_b128(dst + k * LDW + c * 8, src + (size_t)k * 128 + c * 8);
  }
}

// ---------------------------------------------------------------------------
// Small utility kernels
// ---------------------------------------------------------------------------
__global__ void cvt_f32_bf16_kernel(const float* __restrict__ s,
                                    bf16* __restrict__ d, int n) {
  int i = blockIdx.x * blockDim.x + threadIdx.x;
  if (i < n) d[i] = (bf16)s[i];
}

__global__ void zero_f32_kernel(float* __restrict__ p, int n) {
  int i = blockIdx.x * blockDim.x + threadIdx.x;
  if (i < n) p[i] = 0.0f;
}

// ---------------------------------------------------------------------------
// Encoder: hidden = relu(x @ W1 + b1) via VALU (tiny K), then WMMA GEMM2 + LN
// ---------------------------------------------------------------------------
template <int DIN>
__global__ void encoder_kernel(const float* __restrict__ xin,
                               const float* __restrict__ W1f,
                               const float* __restrict__ b1,
                               const bf16* __restrict__ W2,
                               const float* __restrict__ b2,
                               const float* __restrict__ gain,
                               const float* __restrict__ beta,
                               bf16* __restrict__ out, int nrows) {
  extern __shared__ __align__(16) char smem[];
  bf16* Hs = (bf16*)(smem);                 // [128][LDW]
  bf16* W2s = (bf16*)(smem + 34816);        // [128][LDW]
  float* OUTs = (float*)(smem + 69632);     // [128][LDO]
  float* red = (float*)(smem + 137216);     // [512]
  const int tid = threadIdx.x;
  const int tile0 = blockIdx.x * 128;

  stage_weight(W2s, W2, 128, tid);

  for (int o = tid; o < 128 * 128; o += 256) {
    const int r = o >> 7, c = o & 127;
    const int gr = tile0 + r;
    float s = 0.0f;
    if (gr < nrows) {
      s = b1[c];
#pragma unroll
      for (int k = 0; k < DIN; ++k)
        s += xin[(size_t)gr * DIN + k] * W1f[k * 128 + c];
      s = fmaxf(s, 0.0f);
    }
    Hs[r * LDW + c] = (bf16)s;
  }
  wait_async_lds();
  __syncthreads();

  block_gemm_128<128, 1>(Hs, LDW, W2s, LDW, b2, nullptr, 0, OUTs, LDO);
  __syncthreads();

  float mu, rs;
  row_stats(OUTs, red, tid, mu, rs);
  const int row = tid >> 1, half = tid & 1;
  const int gr = tile0 + row;
  if (gr < nrows) {
    for (int c = half * 64; c < half * 64 + 64; ++c) {
      float v = (OUTs[row * LDO + c] - mu) * rs * gain[c] + beta[c];
      out[(size_t)gr * 128 + c] = (bf16)v;
    }
  }
}

// ---------------------------------------------------------------------------
// Edge layer: e_new = LN(MLP(cat(h[dst], h[src], e))) + e; scatter-add to agg
// ---------------------------------------------------------------------------
__global__ void edge_layer_kernel(const int* __restrict__ ei,
                                  const bf16* __restrict__ hbf,
                                  bf16* __restrict__ ebf,
                                  const bf16* __restrict__ W1,
                                  const float* __restrict__ b1,
                                  const bf16* __restrict__ W2,
                                  const float* __restrict__ b2,
                                  const float* __restrict__ gain,
                                  const float* __restrict__ beta,
                                  float* __restrict__ agg) {
  extern __shared__ __align__(16) char smem[];
  bf16* As = (bf16*)(smem);                  // [128][LDA_E]  (100352 B)
  bf16* W1s = (bf16*)(smem + 100352);        // [384][LDW]    (104448 B)
  bf16* W2s = (bf16*)(smem + 204800);        // [128][LDW]    (34816 B)
  bf16* Hs = (bf16*)(smem + 239616);         // [128][LDW]    (34816 B)
  float* OUTs = (float*)(smem);              // alias of As
  int* sIdx = (int*)(smem + 274432);         // [128]
  int* dIdx = (int*)(smem + 274944);         // [128]
  float* red = (float*)(smem + 275456);      // [512]
  const int tid = threadIdx.x;
  const int tile0 = blockIdx.x * 128;

  if (tid < 128) {
    sIdx[tid] = ei[tile0 + tid];            // edge_index[0] (x_j / agg target)
    dIdx[tid] = ei[N_EDGES + tile0 + tid];  // edge_index[1] (x_i)
  }
  __syncthreads();

  stage_weight(W1s, W1, 384, tid);
  stage_weight(W2s, W2, 128, tid);

  // Gather A = cat(h[dst], h[src], e), 16B chunks via async-to-LDS.
  for (int i = tid; i < 128 * 48; i += 256) {
    const int row = i / 48, c = i - row * 48;
    const bf16* src;
    if (c < 16)
      src = hbf + (size_t)dIdx[row] * 128 + (c << 3);
    else if (c < 32)
      src = hbf + (size_t)sIdx[row] * 128 + ((c - 16) << 3);
    else
      src = ebf + (size_t)(tile0 + row) * 128 + ((c - 32) << 3);
    g2l_b128(As + row * LDA_E + c * 8, src);
  }
  wait_async_lds();
  __syncthreads();

  block_gemm_128<384, 0>(As, LDA_E, W1s, LDW, b1, Hs, LDW, nullptr, 0);
  __syncthreads();
  block_gemm_128<128, 1>(Hs, LDW, W2s, LDW, b2, nullptr, 0, OUTs, LDO);
  __syncthreads();

  float mu, rs;
  row_stats(OUTs, red, tid, mu, rs);
  const int row = tid >> 1, half = tid & 1;
  const size_t erow = (size_t)(tile0 + row) * 128;
  const size_t arow = (size_t)sIdx[row] * 128;
  for (int c = half * 64; c < half * 64 + 64; ++c) {
    float v = (OUTs[row * LDO + c] - mu) * rs * gain[c] + beta[c] +
              (float)ebf[erow + c];
    ebf[erow + c] = (bf16)v;
    atomicAdd(&agg[arow + c], v);
  }
}

// ---------------------------------------------------------------------------
// Node layer: h = h + LN(MLP(cat(h, agg)))
// ---------------------------------------------------------------------------
__global__ void node_layer_kernel(bf16* __restrict__ hbf,
                                  const float* __restrict__ agg,
                                  const bf16* __restrict__ W1,
                                  const float* __restrict__ b1,
                                  const bf16* __restrict__ W2,
                                  const float* __restrict__ b2,
                                  const float* __restrict__ gain,
                                  const float* __restrict__ beta, int nrows) {
  extern __shared__ __align__(16) char smem[];
  bf16* As = (bf16*)(smem);                 // [128][LDA_N] (67584 B)
  bf16* W1s = (bf16*)(smem + 67584);        // [256][LDW]   (69632 B)
  bf16* W2s = (bf16*)(smem + 137216);       // [128][LDW]   (34816 B)
  bf16* Hs = (bf16*)(smem + 172032);        // [128][LDW]   (34816 B)
  float* OUTs = (float*)(smem);             // alias of As
  float* red = (float*)(smem + 206848);     // [512]
  const int tid = threadIdx.x;
  const int tile0 = blockIdx.x * 128;

  stage_weight(W1s, W1, 256, tid);
  stage_weight(W2s, W2, 128, tid);

  // Columns 0..127: h rows (async), pad with zeros past nrows.
  for (int i = tid; i < 128 * 16; i += 256) {
    const int row = i >> 4, c = i & 15;
    const int gr = tile0 + row;
    if (gr < nrows) {
      g2l_b128(As + row * LDA_N + c * 8, hbf + (size_t)gr * 128 + (c << 3));
    } else {
      *(uint4*)(As + row * LDA_N + c * 8) = make_uint4(0u, 0u, 0u, 0u);
    }
  }
  // Columns 128..255: agg (f32 -> bf16 convert).
  for (int o = tid; o < 128 * 128; o += 256) {
    const int r = o >> 7, c = o & 127;
    const int gr = tile0 + r;
    float v = (gr < nrows) ? agg[(size_t)gr * 128 + c] : 0.0f;
    As[r * LDA_N + 128 + c] = (bf16)v;
  }
  wait_async_lds();
  __syncthreads();

  block_gemm_128<256, 0>(As, LDA_N, W1s, LDW, b1, Hs, LDW, nullptr, 0);
  __syncthreads();
  block_gemm_128<128, 1>(Hs, LDW, W2s, LDW, b2, nullptr, 0, OUTs, LDO);
  __syncthreads();

  float mu, rs;
  row_stats(OUTs, red, tid, mu, rs);
  const int row = tid >> 1, half = tid & 1;
  const int gr = tile0 + row;
  if (gr < nrows) {
    for (int c = half * 64; c < half * 64 + 64; ++c) {
      float v = (OUTs[row * LDO + c] - mu) * rs * gain[c] + beta[c];
      float hn = (float)hbf[(size_t)gr * 128 + c] + v;
      hbf[(size_t)gr * 128 + c] = (bf16)hn;
    }
  }
}

// ---------------------------------------------------------------------------
// Decoder: out = relu(h @ W1 + b1) @ W2 + b2   (OUT = 2)
// ---------------------------------------------------------------------------
__global__ void decoder_kernel(const bf16* __restrict__ hbf,
                               const bf16* __restrict__ W1,
                               const float* __restrict__ b1,
                               const float* __restrict__ W2f,
                               const float* __restrict__ b2,
                               float* __restrict__ out, int nrows) {
  extern __shared__ __align__(16) char smem[];
  bf16* As = (bf16*)(smem);              // [128][LDA_D] (34816 B)
  bf16* W1s = (bf16*)(smem + 34816);     // [128][LDW]   (34816 B)
  float* OUTs = (float*)(smem + 69632);  // [128][LDO]   (67584 B)
  const int tid = threadIdx.x;
  const int tile0 = blockIdx.x * 128;

  stage_weight(W1s, W1, 128, tid);
  for (int i = tid; i < 128 * 16; i += 256) {
    const int row = i >> 4, c = i & 15;
    const int gr = tile0 + row;
    if (gr < nrows) {
      g2l_b128(As + row * LDA_D + c * 8, hbf + (size_t)gr * 128 + (c << 3));
    } else {
      *(uint4*)(As + row * LDA_D + c * 8) = make_uint4(0u, 0u, 0u, 0u);
    }
  }
  wait_async_lds();
  __syncthreads();

  block_gemm_128<128, 2>(As, LDA_D, W1s, LDW, b1, nullptr, 0, OUTs, LDO);
  __syncthreads();

  if (tid < 256) {
    const int r = tid >> 1, oc = tid & 1;
    const int gr = tile0 + r;
    if (gr < nrows) {
      float s = b2[oc];
#pragma unroll 8
      for (int k = 0; k < 128; ++k) s += OUTs[r * LDO + k] * W2f[k * 2 + oc];
      out[(size_t)gr * 2 + oc] = s;
    }
  }
}

// ---------------------------------------------------------------------------
// Host orchestration
// ---------------------------------------------------------------------------
extern "C" void kernel_launch(void* const* d_in, const int* in_sizes, int n_in,
                              void* d_out, int out_size, void* d_ws,
                              size_t ws_size, hipStream_t stream) {
  (void)in_sizes; (void)n_in; (void)out_size; (void)ws_size;

  // Input order: recursive insertion-order flatten of setup_inputs() dict.
  const float* x = (const float*)d_in[0];
  const int* ei = (const int*)d_in[1];
  const float* eattr = (const float*)d_in[2];
  const float* neW1 = (const float*)d_in[3];
  const float* neb1 = (const float*)d_in[4];
  const float* neW2 = (const float*)d_in[5];
  const float* neb2 = (const float*)d_in[6];
  const float* neg_ = (const float*)d_in[7];
  const float* nebt = (const float*)d_in[8];
  const float* eeW1 = (const float*)d_in[9];
  const float* eeb1 = (const float*)d_in[10];
  const float* eeW2 = (const float*)d_in[11];
  const float* eeb2 = (const float*)d_in[12];
  const float* eeg = (const float*)d_in[13];
  const float* eebt = (const float*)d_in[14];
  const float* peW1 = (const float*)d_in[15];  // [8,384,128]
  const float* peb1 = (const float*)d_in[16];
  const float* peW2 = (const float*)d_in[17];  // [8,128,128]
  const float* peb2 = (const float*)d_in[18];
  const float* peg = (const float*)d_in[19];
  const float* pebt = (const float*)d_in[20];
  const float* pnW1 = (const float*)d_in[21];  // [8,256,128]
  const float* pnb1 = (const float*)d_in[22];
  const float* pnW2 = (const float*)d_in[23];
  const float* pnb2 = (const float*)d_in[24];
  const float* png = (const float*)d_in[25];
  const float* pnbt = (const float*)d_in[26];
  const float* dW1 = (const float*)d_in[27];
  const float* db1 = (const float*)d_in[28];
  const float* dW2 = (const float*)d_in[29];
  const float* db2 = (const float*)d_in[30];

  char* ws = (char*)d_ws;
  size_t off = 0;
  auto take = [&](size_t bytes) -> char* {
    char* p = ws + off;
    off += (bytes + 255) & ~(size_t)255;
    return p;
  };
  bf16* hbf = (bf16*)take((size_t)N_NODES * 128 * 2);
  bf16* ebf = (bf16*)take((size_t)N_EDGES * 128 * 2);
  float* agg = (float*)take((size_t)N_NODES * 128 * 4);
  bf16* p_neW2 = (bf16*)take(16384 * 2);
  bf16* p_eeW2 = (bf16*)take(16384 * 2);
  bf16* p_peW1 = (bf16*)take((size_t)8 * 384 * 128 * 2);
  bf16* p_peW2 = (bf16*)take((size_t)8 * 16384 * 2);
  bf16* p_pnW1 = (bf16*)take((size_t)8 * 256 * 128 * 2);
  bf16* p_pnW2 = (bf16*)take((size_t)8 * 16384 * 2);
  bf16* p_dW1 = (bf16*)take(16384 * 2);

  auto cvt = [&](const float* s, bf16* d, int n) {
    cvt_f32_bf16_kernel<<<(n + 255) / 256, 256, 0, stream>>>(s, d, n);
  };
  cvt(neW2, p_neW2, 16384);
  cvt(eeW2, p_eeW2, 16384);
  cvt(peW1, p_peW1, 8 * 384 * 128);
  cvt(peW2, p_peW2, 8 * 16384);
  cvt(pnW1, p_pnW1, 8 * 256 * 128);
  cvt(pnW2, p_pnW2, 8 * 16384);
  cvt(dW1, p_dW1, 16384);

  const int nodeBlocks = (N_NODES + 127) / 128;  // 391
  const int edgeBlocks = N_EDGES / 128;          // 6250
  constexpr size_t ENC_SMEM = 139264;
  constexpr size_t EDGE_SMEM = 277504;
  constexpr size_t NODE_SMEM = 208896;
  constexpr size_t DEC_SMEM = 137216;

  encoder_kernel<11><<<nodeBlocks, 256, ENC_SMEM, stream>>>(
      x, neW1, neb1, p_neW2, neb2, neg_, nebt, hbf, N_NODES);
  encoder_kernel<3><<<edgeBlocks, 256, ENC_SMEM, stream>>>(
      eattr, eeW1, eeb1, p_eeW2, eeb2, eeg, eebt, ebf, N_EDGES);

  for (int l = 0; l < DEPTH; ++l) {
    zero_f32_kernel<<<(N_NODES * 128 + 255) / 256, 256, 0, stream>>>(
        agg, N_NODES * 128);
    edge_layer_kernel<<<edgeBlocks, 256, EDGE_SMEM, stream>>>(
        ei, hbf, ebf, p_peW1 + (size_t)l * 384 * 128, peb1 + l * 128,
        p_peW2 + (size_t)l * 16384, peb2 + l * 128, peg + l * 128,
        pebt + l * 128, agg);
    node_layer_kernel<<<nodeBlocks, 256, NODE_SMEM, stream>>>(
        hbf, agg, p_pnW1 + (size_t)l * 256 * 128, pnb1 + l * 128,
        p_pnW2 + (size_t)l * 16384, pnb2 + l * 128, png + l * 128,
        pnbt + l * 128, N_NODES);
  }

  decoder_kernel<<<nodeBlocks, 256, DEC_SMEM, stream>>>(
      hbf, p_dW1, db1, dW2, db2, (float*)d_out, N_NODES);
}
